// HomographyNet_14491219657117
// MI455X (gfx1250) — compile-verified, matrix operand order checked by reading, not run on hardware
//
#include <hip/hip_runtime.h>
#include <hip/hip_bf16.h>

// ---------------------------------------------------------------------------
// HomographyNet pipeline for gfx1250 (MI455X), wave32 + WMMA f16->f32.
//   correlation  -> banded GEMM via v_wmma_f32_16x16x32_f16
//   conv3x3      -> implicit GEMM via WMMA (BN+ReLU fused epilogue)
//   FC layers    -> WMMA GEMM (batch padded 8->16 rows)
//   DLT / inv3x3 -> per-lane scalar solve
//   warps        -> bilinear gather kernels (memory bound)
// ---------------------------------------------------------------------------

typedef __attribute__((ext_vector_type(16))) _Float16 v16h;
typedef __attribute__((ext_vector_type(8)))  _Float16 v8h;
typedef __attribute__((ext_vector_type(8)))  float    v8f;

#define CDIV(a,b) (((a)+(b)-1)/(b))

__device__ __forceinline__ v8f zero8f() {
  v8f a;
#pragma unroll
  for (int i = 0; i < 8; ++i) a[i] = 0.f;
  return a;
}
__device__ __forceinline__ v16h zero16h() {
  v16h a;
#pragma unroll
  for (int i = 0; i < 16; ++i) a[i] = (_Float16)0.f;
  return a;
}
// 16-bit A/B fragment (16x32 / 32x16): lane<16 holds K {0..7,16..23},
// lane>=16 holds K {8..15,24..31}. p points at this lane's row/col, k0=0.
__device__ __forceinline__ v16h frag_load(const _Float16* p, int koff) {
  v8h lo = *(const v8h*)(p + koff);
  v8h hi = *(const v8h*)(p + 16 + koff);
  v16h f;
#pragma unroll
  for (int i = 0; i < 8; ++i) { f[i] = lo[i]; f[8 + i] = hi[i]; }
  return f;
}
__device__ __forceinline__ v8f wmma16(v16h a, v16h b, v8f c) {
  return __builtin_amdgcn_wmma_f32_16x16x32_f16(false, a, false, b,
                                                (short)0, c, false, false);
}

// ---------------------------------------------------------------------------
// utility kernels
// ---------------------------------------------------------------------------
__global__ void zero_h(_Float16* p, int n) {
  int i = blockIdx.x * blockDim.x + threadIdx.x;
  if (i < n) p[i] = (_Float16)0.f;
}

__global__ void cast_pad_h(const float* __restrict__ s, _Float16* __restrict__ d,
                           int nvalid, int ntotal) {
  int i = blockIdx.x * blockDim.x + threadIdx.x;
  if (i >= ntotal) return;
  d[i] = (i < nvalid) ? (_Float16)s[i] : (_Float16)0.f;
}

// conv weights (Cout,Cin,3,3) f32 -> f16 [tap][co][ciPad], zero padded ci.
__global__ void pack_conv_w(const float* __restrict__ w, _Float16* __restrict__ wp,
                            int Cout, int Cin, int CinPad) {
  int i = blockIdx.x * blockDim.x + threadIdx.x;
  int total = 9 * Cout * CinPad;
  if (i >= total) return;
  int ci = i % CinPad;
  int rest = i / CinPad;
  int co = rest % Cout;
  int t = rest / Cout;
  float v = (ci < Cin) ? w[(co * Cin + ci) * 9 + t] : 0.f;
  wp[i] = (_Float16)v;
}

// (B,C,H,W) f32 -> l2-normalized pixel-major f16 [b][y][x][c]
__global__ void norm_pm(const float* __restrict__ in, _Float16* __restrict__ out,
                        int C, int HW) {
  int pix = blockIdx.x;                 // b*HW + (y*W+x)
  int c = threadIdx.x;
  int b = pix / HW, p = pix % HW;
  float v = in[((size_t)b * C + c) * HW + p];
  __shared__ float red[256];
  red[c] = v * v;
  __syncthreads();
  for (int s = blockDim.x >> 1; s > 0; s >>= 1) {
    if (c < s) red[c] += red[c + s];
    __syncthreads();
  }
  float sc = 1.f / fmaxf(sqrtf(red[0]), 1e-12f);
  out[(size_t)pix * C + c] = (_Float16)(v * sc);
}

// ---------------------------------------------------------------------------
// correlation as banded GEMM.  corr[b, dy*9+dx, y, x] = <fa[.,y,x], fb[.,y+dy-4,x+dx-4]>
// stored pixel-major f16 [b][y][x][96] (81 valid channels, rest zero).
// One wave per (b, y, 16-pixel M tile).
// ---------------------------------------------------------------------------
template <int C, int W>
__global__ void corr_wmma(const _Float16* __restrict__ fa,
                          const _Float16* __restrict__ fb,
                          _Float16* __restrict__ corr) {
  constexpr int MT = W / 16;
  constexpr int NK = C / 32;
  int blk = blockIdx.x;
  int mt = blk % MT;
  int y = (blk / MT) % W;
  int b = blk / (MT * W);
  int lane = threadIdx.x;
  int m0 = mt * 16;
  int row = m0 + (lane & 15);
  int koff = (lane & 16) ? 8 : 0;
  int mhalf = (lane & 16) ? 8 : 0;

  v16h afr[NK];
  const _Float16* abase = fa + (((size_t)(b * W + y)) * W + row) * C;
#pragma unroll
  for (int k = 0; k < NK; ++k) afr[k] = frag_load(abase + k * 32, koff);

  for (int dy = 0; dy < 9; ++dy) {
    int yb = y + dy - 4;
    if (yb < 0 || yb >= W) continue;   // vertical zero-pad: buffer pre-zeroed
#pragma unroll
    for (int nt = 0; nt < 2; ++nt) {
      int n0 = m0 - 4 + nt * 16;       // two tiles cover the +/-4 band
      int col = n0 + (lane & 15);
      bool cv = (col >= 0 && col < W);
      const _Float16* bbase =
          fb + (((size_t)(b * W + yb)) * W + (cv ? col : 0)) * C;
      v8f acc = zero8f();
#pragma unroll
      for (int k = 0; k < NK; ++k) {
        v16h bf = cv ? frag_load(bbase + k * 32, koff) : zero16h();
        acc = wmma16(afr[k], bf, acc);
      }
#pragma unroll
      for (int r = 0; r < 8; ++r) {
        int x = m0 + r + mhalf;
        int xc = n0 + (lane & 15);
        int dxi = xc - x + 4;
        if (dxi >= 0 && dxi < 9 && xc >= 0 && xc < W) {
          corr[(((size_t)(b * W + y)) * W + x) * 96 + dy * 9 + dxi] =
              (_Float16)acc[r];
        }
      }
    }
  }
}

// ---------------------------------------------------------------------------
// 3x3 conv, implicit GEMM.  x: pixel-major f16 [b][y][x][CIN] (zero padded
// channels).  wp: f16 [tap][co][CIN].  Fused bias + BN(eval) + ReLU, output
// pixel-major f16 [b][y][x][COUT].  One wave per (b,y,Mtile,Ntile).
// ---------------------------------------------------------------------------
template <int CIN, int COUT, int W>
__global__ void conv3x3_wmma(const _Float16* __restrict__ x,
                             const _Float16* __restrict__ wp,
                             const float* __restrict__ bias,
                             const float* __restrict__ g,
                             const float* __restrict__ be,
                             const float* __restrict__ mu,
                             const float* __restrict__ var,
                             _Float16* __restrict__ out) {
  constexpr int MT = W / 16, NT = COUT / 16, NK = CIN / 32;
  int blk = blockIdx.x;
  int nt = blk % NT;
  int mt = (blk / NT) % MT;
  int y = (blk / (NT * MT)) % W;
  int b = blk / (NT * MT * W);
  int lane = threadIdx.x;
  int m0 = mt * 16, n0 = nt * 16;
  int koff = (lane & 16) ? 8 : 0;
  int mhalf = (lane & 16) ? 8 : 0;
  int co = n0 + (lane & 15);

  v8f acc = zero8f();
#pragma unroll
  for (int t = 0; t < 9; ++t) {
    int ky = t / 3, kx = t % 3;
    int ys = y + ky - 1;
    if (ys < 0 || ys >= W) continue;
    int xs = m0 + (lane & 15) + kx - 1;
    bool av = (xs >= 0 && xs < W);
    const _Float16* ab = x + (((size_t)(b * W + ys)) * W + (av ? xs : 0)) * CIN;
    const _Float16* bb = wp + ((size_t)(t * COUT) + co) * CIN;
    if (t < 8) __builtin_prefetch(wp + ((size_t)((t + 1) * COUT) + co) * CIN, 0, 1);
#pragma unroll
    for (int k = 0; k < NK; ++k) {
      v16h af = av ? frag_load(ab + k * 32, koff) : zero16h();
      v16h bf = frag_load(bb + k * 32, koff);
      acc = wmma16(af, bf, acc);
    }
  }
  float s = g[co] * rsqrtf(var[co] + 1e-5f);
  float b0 = bias[co], m0v = mu[co], be0 = be[co];
#pragma unroll
  for (int r = 0; r < 8; ++r) {
    int xo = m0 + r + mhalf;
    float v = (acc[r] + b0 - m0v) * s + be0;
    v = fmaxf(v, 0.f);
    out[(((size_t)(b * W + y)) * W + xo) * COUT + co] = (_Float16)v;
  }
}

// AdaptiveAvgPool2d((4,4)) on pixel-major f16 [b][y][x][64] -> f16 [16][1024]
// (rows >= 8 stay zero; flat index c*16 + gy*4 + gx matches the reference).
__global__ void pool44(const _Float16* __restrict__ x, _Float16* __restrict__ out,
                       int W) {
  int i = blockIdx.x * blockDim.x + threadIdx.x;
  if (i >= 8 * 64 * 16) return;
  int gx = i & 3, gy = (i >> 2) & 3, c = (i >> 4) & 63, b = i >> 10;
  int bs = W / 4;
  float s = 0.f;
  for (int yy = 0; yy < bs; ++yy)
    for (int xx = 0; xx < bs; ++xx) {
      int y = gy * bs + yy, xp = gx * bs + xx;
      s += (float)x[(((size_t)(b * W + y)) * W + xp) * 64 + c];
    }
  s /= (float)(bs * bs);
  out[b * 1024 + c * 16 + gy * 4 + gx] = (_Float16)s;
}

// FC GEMM: A f16 [16][K] (rows>=8 zero), Wt f16 [N][K].  mode=1: ReLU -> f16
// outh[m*1024+j];  mode=0: f32 delta outf[m*8+j] (j<8).  One wave per N-tile.
__global__ void fc_wmma(const _Float16* __restrict__ A,
                        const _Float16* __restrict__ Wt,
                        const float* __restrict__ bias, float* __restrict__ outf,
                        _Float16* __restrict__ outh, int K, int mode) {
  int n0 = blockIdx.x * 16;
  int lane = threadIdx.x;
  int koff = (lane & 16) ? 8 : 0;
  int mhalf = (lane & 16) ? 8 : 0;
  const _Float16* ab = A + (size_t)(lane & 15) * K;
  const _Float16* bb = Wt + (size_t)(n0 + (lane & 15)) * K;
  v8f acc = zero8f();
  for (int k = 0; k < K; k += 32) {
    v16h af = frag_load(ab + k, koff);
    v16h bf = frag_load(bb + k, koff);
    acc = wmma16(af, bf, acc);
  }
#pragma unroll
  for (int r = 0; r < 8; ++r) {
    int m = r + mhalf;
    int j = n0 + (lane & 15);
    if (m < 8) {
      if (mode == 1) {
        float v = fmaxf(acc[r] + bias[j], 0.f);
        outh[m * 1024 + j] = (_Float16)v;
      } else if (j < 8) {
        outf[m * 8 + j] = acc[r] + bias[j];
      }
    }
  }
}

// ---------------------------------------------------------------------------
// tensor DLT: per-lane 8x8 solve (partial pivoting) + 3x3 inverse.
// ---------------------------------------------------------------------------
__device__ __forceinline__ float n2n(float v) {  // nan_to_num(nan/inf -> 0)
  return (fabsf(v) <= 1e30f) ? v : 0.f;
}

__global__ void dlt_kernel(const float* __restrict__ delta,
                           float* __restrict__ Hout, float* __restrict__ Hinv) {
  int b = threadIdx.x;
  if (b >= 8) return;
  float d[8];
#pragma unroll
  for (int i = 0; i < 8; ++i)
    d[i] = fminf(fmaxf(n2n(delta[b * 8 + i]), -2.f), 2.f);
  const float sx[4] = {-1.f, 1.f, 1.f, -1.f};
  const float sy[4] = {-1.f, -1.f, 1.f, 1.f};
  float A[8][9];
#pragma unroll
  for (int p = 0; p < 4; ++p) {
    float xs = sx[p], ys = sy[p];
    float xd = xs + d[2 * p], yd = ys + d[2 * p + 1];
    float* r1 = A[p];
    float* r2 = A[p + 4];
    r1[0]=xs; r1[1]=ys; r1[2]=1.f; r1[3]=0.f; r1[4]=0.f; r1[5]=0.f;
    r1[6]=-xd*xs; r1[7]=-xd*ys; r1[8]=xd;
    r2[0]=0.f; r2[1]=0.f; r2[2]=0.f; r2[3]=xs; r2[4]=ys; r2[5]=1.f;
    r2[6]=-yd*xs; r2[7]=-yd*ys; r2[8]=yd;
  }
  for (int c = 0; c < 8; ++c) {
    int piv = c;
    float best = fabsf(A[c][c]);
    for (int r = c + 1; r < 8; ++r) {
      float a = fabsf(A[r][c]);
      if (a > best) { best = a; piv = r; }
    }
    if (piv != c)
      for (int k = c; k < 9; ++k) { float t = A[c][k]; A[c][k] = A[piv][k]; A[piv][k] = t; }
    float inv = 1.f / A[c][c];
    for (int r = c + 1; r < 8; ++r) {
      float f = A[r][c] * inv;
      for (int k = c; k < 9; ++k) A[r][k] -= f * A[c][k];
    }
  }
  float h[9];
  for (int r = 7; r >= 0; --r) {
    float s = A[r][8];
    for (int k = r + 1; k < 8; ++k) s -= A[r][k] * h[k];
    h[r] = s / A[r][r];
  }
  h[8] = 1.f;
#pragma unroll
  for (int i = 0; i < 9; ++i) { h[i] = n2n(h[i]); Hout[b * 9 + i] = h[i]; }
  // 3x3 inverse (adjugate / det) of the nan_to_num'ed H
  float a0=h[0],a1=h[1],a2=h[2],a3=h[3],a4=h[4],a5=h[5],a6=h[6],a7=h[7],a8=h[8];
  float c0=a4*a8-a5*a7, c1=a2*a7-a1*a8, c2=a1*a5-a2*a4;
  float c3=a5*a6-a3*a8, c4=a0*a8-a2*a6, c5=a2*a3-a0*a5;
  float c6=a3*a7-a4*a6, c7=a1*a6-a0*a7, c8=a0*a4-a1*a3;
  float idet = 1.f / (a0*c0 + a1*c3 + a2*c6);
  Hinv[b*9+0]=c0*idet; Hinv[b*9+1]=c1*idet; Hinv[b*9+2]=c2*idet;
  Hinv[b*9+3]=c3*idet; Hinv[b*9+4]=c4*idet; Hinv[b*9+5]=c5*idet;
  Hinv[b*9+6]=c6*idet; Hinv[b*9+7]=c7*idet; Hinv[b*9+8]=c8*idet;
}

// ---------------------------------------------------------------------------
// warp fb_8 with Hinv1 + l2norm over channels -> pixel-major f16 [b][y][x][128]
// ---------------------------------------------------------------------------
__global__ void warp_feat(const float* __restrict__ fb8,
                          const float* __restrict__ Hinv,
                          _Float16* __restrict__ out) {
  const int W = 64, C = 128;
  int pix = blockIdx.x;
  int x = pix % W, y = (pix / W) % W, b = pix / (W * W);
  int c = threadIdx.x;
  const float* h = Hinv + b * 9;
  float gx = -1.f + 2.f * x / (float)(W - 1);
  float gy = -1.f + 2.f * y / (float)(W - 1);
  float mx = h[0]*gx + h[1]*gy + h[2];
  float my = h[3]*gx + h[4]*gy + h[5];
  float mz = h[6]*gx + h[7]*gy + h[8] + 1e-8f;
  float px = (mx / mz + 1.f) * 0.5f * (W - 1);
  float py = (my / mz + 1.f) * 0.5f * (W - 1);
  float x0f = floorf(px), y0f = floorf(py);
  float wx = px - x0f, wy = py - y0f;
  int x0 = (int)x0f, y0 = (int)y0f;
  const float* img = fb8 + ((size_t)b * C + c) * W * W;
  float v00 = (x0 >= 0 && x0 < W && y0 >= 0 && y0 < W) ? img[y0 * W + x0] : 0.f;
  float v10 = (x0 + 1 >= 0 && x0 + 1 < W && y0 >= 0 && y0 < W) ? img[y0 * W + x0 + 1] : 0.f;
  float v01 = (x0 >= 0 && x0 < W && y0 + 1 >= 0 && y0 + 1 < W) ? img[(y0 + 1) * W + x0] : 0.f;
  float v11 = (x0 + 1 >= 0 && x0 + 1 < W && y0 + 1 >= 0 && y0 + 1 < W) ? img[(y0 + 1) * W + x0 + 1] : 0.f;
  float v = v00 * (1 - wx) * (1 - wy) + v10 * wx * (1 - wy) +
            v01 * (1 - wx) * wy + v11 * wx * wy;
  __shared__ float red[128];
  red[c] = v * v;
  __syncthreads();
  for (int s = blockDim.x >> 1; s > 0; s >>= 1) {
    if (c < s) red[c] += red[c + s];
    __syncthreads();
  }
  float sc = 1.f / fmaxf(sqrtf(red[0]), 1e-12f);
  out[(size_t)pix * C + c] = (_Float16)(v * sc);
}

// ---------------------------------------------------------------------------
// final warp: img_a, img_b (B,3,512,512) + mask, all written to d_out
// ---------------------------------------------------------------------------
__global__ void warp_img(const float* __restrict__ ia, const float* __restrict__ ib,
                         const float* __restrict__ Hinv, float* __restrict__ wa,
                         float* __restrict__ wb, float* __restrict__ mask) {
  const int W = 512;
  int i = blockIdx.x * blockDim.x + threadIdx.x;
  if (i >= 8 * W * W) return;
  int x = i % W, y = (i / W) % W, b = i / (W * W);
  const float* h = Hinv + b * 9;
  float gx = -1.f + 2.f * x / (float)(W - 1);
  float gy = -1.f + 2.f * y / (float)(W - 1);
  float mx = h[0]*gx + h[1]*gy + h[2];
  float my = h[3]*gx + h[4]*gy + h[5];
  float mz = h[6]*gx + h[7]*gy + h[8] + 1e-8f;
  float px = (mx / mz + 1.f) * 0.5f * (W - 1);
  float py = (my / mz + 1.f) * 0.5f * (W - 1);
  float x0f = floorf(px), y0f = floorf(py);
  float wx = px - x0f, wy = py - y0f;
  int x0 = (int)x0f, y0 = (int)y0f;
  bool i00 = (x0 >= 0 && x0 < W && y0 >= 0 && y0 < W);
  bool i10 = (x0 + 1 >= 0 && x0 + 1 < W && y0 >= 0 && y0 < W);
  bool i01 = (x0 >= 0 && x0 < W && y0 + 1 >= 0 && y0 + 1 < W);
  bool i11 = (x0 + 1 >= 0 && x0 + 1 < W && y0 + 1 >= 0 && y0 + 1 < W);
  float w00 = (1 - wx) * (1 - wy), w10 = wx * (1 - wy);
  float w01 = (1 - wx) * wy, w11 = wx * wy;
  float msum = (i00 ? w00 : 0.f) + (i10 ? w10 : 0.f) + (i01 ? w01 : 0.f) + (i11 ? w11 : 0.f);
  size_t pidx = (size_t)b * W * W + (size_t)y * W + x;
  mask[pidx] = (msum > 0.5f) ? 1.f : 0.f;
#pragma unroll
  for (int c = 0; c < 3; ++c) {
    const float* pa = ia + ((size_t)(b * 3 + c)) * W * W;
    const float* pb = ib + ((size_t)(b * 3 + c)) * W * W;
    float a00 = i00 ? pa[y0 * W + x0] : 0.f;
    float a10 = i10 ? pa[y0 * W + x0 + 1] : 0.f;
    float a01 = i01 ? pa[(y0 + 1) * W + x0] : 0.f;
    float a11 = i11 ? pa[(y0 + 1) * W + x0 + 1] : 0.f;
    float b00 = i00 ? pb[y0 * W + x0] : 0.f;
    float b10 = i10 ? pb[y0 * W + x0 + 1] : 0.f;
    float b01 = i01 ? pb[(y0 + 1) * W + x0] : 0.f;
    float b11 = i11 ? pb[(y0 + 1) * W + x0 + 1] : 0.f;
    size_t oidx = ((size_t)(b * 3 + c)) * W * W + (size_t)y * W + x;
    wa[oidx] = a00 * w00 + a10 * w10 + a01 * w01 + a11 * w11;
    wb[oidx] = b00 * w00 + b10 * w10 + b01 * w01 + b11 * w11;
  }
}

// ---------------------------------------------------------------------------
// launch
// ---------------------------------------------------------------------------
extern "C" void kernel_launch(void* const* d_in, const int* in_sizes, int n_in,
                              void* d_out, int out_size, void* d_ws, size_t ws_size,
                              hipStream_t stream) {
  (void)in_sizes; (void)n_in; (void)out_size; (void)ws_size;
  const float* img_a = (const float*)d_in[0];
  const float* img_b = (const float*)d_in[1];
  const float* fa_8  = (const float*)d_in[2];
  const float* fb_8  = (const float*)d_in[3];
  const float* fa_16 = (const float*)d_in[4];
  const float* fb_16 = (const float*)d_in[5];
  // param dict order: w1 b1 g1 be1 m1 v1 w2 b2 g2 be2 m2 v2 fw1 fb1 fw2 fb2
  const int PC = 6, PF = 22;
  auto P = [&](int base, int k) { return (const float*)d_in[base + k]; };

  float* out = (float*)d_out;
  float* H1_out = out;                  // 72
  float* H2_out = out + 72;             // 72
  float* wa_out = out + 144;            // 8*3*512*512
  float* wb_out = wa_out + 8 * 3 * 512 * 512;
  float* mk_out = wb_out + 8 * 3 * 512 * 512;

  // workspace layout (256B aligned), ~51 MB total
  char* wp_ = (char*)d_ws;
  auto alloc = [&](size_t bytes) {
    char* p = wp_;
    wp_ += (bytes + 255) & ~(size_t)255;
    return p;
  };
  _Float16* na16   = (_Float16*)alloc((size_t)8 * 32 * 32 * 256 * 2);
  _Float16* nb16   = (_Float16*)alloc((size_t)8 * 32 * 32 * 256 * 2);
  _Float16* na8    = (_Float16*)alloc((size_t)8 * 64 * 64 * 128 * 2);
  _Float16* nwb8   = (_Float16*)alloc((size_t)8 * 64 * 64 * 128 * 2);
  _Float16* corr_c = (_Float16*)alloc((size_t)8 * 32 * 32 * 96 * 2);
  _Float16* corr_f = (_Float16*)alloc((size_t)8 * 64 * 64 * 96 * 2);
  _Float16* c1c    = (_Float16*)alloc((size_t)8 * 32 * 32 * 128 * 2);
  _Float16* c2c    = (_Float16*)alloc((size_t)8 * 32 * 32 * 64 * 2);
  _Float16* c1f    = (_Float16*)alloc((size_t)8 * 64 * 64 * 128 * 2);
  _Float16* c2f    = (_Float16*)alloc((size_t)8 * 64 * 64 * 64 * 2);
  _Float16* wpk1   = (_Float16*)alloc((size_t)9 * 128 * 96 * 2);
  _Float16* wpk2   = (_Float16*)alloc((size_t)9 * 64 * 128 * 2);
  _Float16* fw1h   = (_Float16*)alloc((size_t)1024 * 1024 * 2);
  _Float16* fw2h   = (_Float16*)alloc((size_t)16 * 1024 * 2);
  _Float16* pooled = (_Float16*)alloc((size_t)16 * 1024 * 2);
  _Float16* fc1h   = (_Float16*)alloc((size_t)16 * 1024 * 2);
  float*    delta  = (float*)alloc(8 * 8 * 4);
  float*    Hinv1  = (float*)alloc(8 * 9 * 4);
  float*    Hinv2  = (float*)alloc(8 * 9 * 4);

  // ---- normalize features (pixel-major f16) ----
  norm_pm<<<8 * 32 * 32, 256, 0, stream>>>(fa_16, na16, 256, 32 * 32);
  norm_pm<<<8 * 32 * 32, 256, 0, stream>>>(fb_16, nb16, 256, 32 * 32);
  norm_pm<<<8 * 64 * 64, 128, 0, stream>>>(fa_8, na8, 128, 64 * 64);

  // ---- zero padded / partially-written buffers ----
  zero_h<<<CDIV(8 * 32 * 32 * 96, 256), 256, 0, stream>>>(corr_c, 8 * 32 * 32 * 96);
  zero_h<<<CDIV(8 * 64 * 64 * 96, 256), 256, 0, stream>>>(corr_f, 8 * 64 * 64 * 96);
  zero_h<<<CDIV(16 * 1024, 256), 256, 0, stream>>>(pooled, 16 * 1024);
  zero_h<<<CDIV(16 * 1024, 256), 256, 0, stream>>>(fc1h, 16 * 1024);

  // ======================= coarse stage (32x32, C=256) ======================
  pack_conv_w<<<CDIV(9 * 128 * 96, 256), 256, 0, stream>>>(P(PC, 0), wpk1, 128, 81, 96);
  pack_conv_w<<<CDIV(9 * 64 * 128, 256), 256, 0, stream>>>(P(PC, 6), wpk2, 64, 128, 128);
  cast_pad_h<<<CDIV(1024 * 1024, 256), 256, 0, stream>>>(P(PC, 12), fw1h, 1024 * 1024, 1024 * 1024);
  cast_pad_h<<<CDIV(16 * 1024, 256), 256, 0, stream>>>(P(PC, 14), fw2h, 8 * 1024, 16 * 1024);

  corr_wmma<256, 32><<<8 * 32 * 2, 32, 0, stream>>>(na16, nb16, corr_c);
  conv3x3_wmma<96, 128, 32><<<8 * 32 * 2 * 8, 32, 0, stream>>>(
      corr_c, wpk1, P(PC, 1), P(PC, 2), P(PC, 3), P(PC, 4), P(PC, 5), c1c);
  conv3x3_wmma<128, 64, 32><<<8 * 32 * 2 * 4, 32, 0, stream>>>(
      c1c, wpk2, P(PC, 7), P(PC, 8), P(PC, 9), P(PC, 10), P(PC, 11), c2c);
  pool44<<<CDIV(8 * 64 * 16, 256), 256, 0, stream>>>(c2c, pooled, 32);
  fc_wmma<<<64, 32, 0, stream>>>(pooled, fw1h, P(PC, 13), nullptr, fc1h, 1024, 1);
  fc_wmma<<<1, 32, 0, stream>>>(fc1h, fw2h, P(PC, 15), delta, nullptr, 1024, 0);
  dlt_kernel<<<1, 32, 0, stream>>>(delta, H1_out, Hinv1);

  // ---- warp fb_8 with H1^-1, re-normalize ----
  warp_feat<<<8 * 64 * 64, 128, 0, stream>>>(fb_8, Hinv1, nwb8);

  // ======================== fine stage (64x64, C=128) =======================
  pack_conv_w<<<CDIV(9 * 128 * 96, 256), 256, 0, stream>>>(P(PF, 0), wpk1, 128, 81, 96);
  pack_conv_w<<<CDIV(9 * 64 * 128, 256), 256, 0, stream>>>(P(PF, 6), wpk2, 64, 128, 128);
  cast_pad_h<<<CDIV(1024 * 1024, 256), 256, 0, stream>>>(P(PF, 12), fw1h, 1024 * 1024, 1024 * 1024);
  cast_pad_h<<<CDIV(16 * 1024, 256), 256, 0, stream>>>(P(PF, 14), fw2h, 8 * 1024, 16 * 1024);

  corr_wmma<128, 64><<<8 * 64 * 4, 32, 0, stream>>>(na8, nwb8, corr_f);
  conv3x3_wmma<96, 128, 64><<<8 * 64 * 4 * 8, 32, 0, stream>>>(
      corr_f, wpk1, P(PF, 1), P(PF, 2), P(PF, 3), P(PF, 4), P(PF, 5), c1f);
  conv3x3_wmma<128, 64, 64><<<8 * 64 * 4 * 4, 32, 0, stream>>>(
      c1f, wpk2, P(PF, 7), P(PF, 8), P(PF, 9), P(PF, 10), P(PF, 11), c2f);
  pool44<<<CDIV(8 * 64 * 16, 256), 256, 0, stream>>>(c2f, pooled, 64);
  fc_wmma<<<64, 32, 0, stream>>>(pooled, fw1h, P(PF, 13), nullptr, fc1h, 1024, 1);
  fc_wmma<<<1, 32, 0, stream>>>(fc1h, fw2h, P(PF, 15), delta, nullptr, 1024, 0);
  dlt_kernel<<<1, 32, 0, stream>>>(delta, H2_out, Hinv2);

  // ---- final image warps + mask ----
  warp_img<<<CDIV(8 * 512 * 512, 256), 256, 0, stream>>>(img_a, img_b, Hinv2,
                                                         wa_out, wb_out, mk_out);
}
